// RTB_20289425506896
// MI455X (gfx1250) — matrix operand order repaired and checked by reading, not compiled, not run on hardware
//
#include <hip/hip_runtime.h>
#include <hip/hip_bf16.h>

// ---------------------------------------------------------------------------
// Dual-attention block for MI455X (gfx1250, wave32, WMMA).
// All heavy math goes through one templated bf16 WMMA GEMM kernel:
//   v_wmma_f32_16x16x32_bf16, f32 accumulate.
// Block tile 128x128x32, 256 threads (8 waves), wave subtile 32x64
// (2x4 fragments) -> 8 WMMAs per wave per K-step, 64 per block.
// Double-buffered LDS: next K-tile global loads overlap WMMA execution.
// Epilogue (bias / residual / f32-out / bf16-out) selected at compile time.
// Shapes fixed by the reference: B=2, C=256, H=W=64, HW=4096, K9=2304.
// Every GEMM dim here is a multiple of 128 (M,N) / 32 (K): no bounds checks.
// ---------------------------------------------------------------------------

#define CC   256
#define HH   64
#define WW   64
#define HW   4096      // 64*64
#define K9   2304      // 256*9

typedef __attribute__((ext_vector_type(16))) __bf16 v16bf;
typedef __attribute__((ext_vector_type(8)))  float  v8f;

static __device__ __forceinline__ __bf16 f2bf(float f) {
    union { float f; unsigned u; } c; c.f = f;
    unsigned r = (c.u + 0x7FFFu + ((c.u >> 16) & 1u)) >> 16;   // RNE
    union { unsigned short s; __bf16 b; } o; o.s = (unsigned short)r;
    return o.b;
}

union V8BF  { uint4 u; __bf16 h[8]; };
union V16BF { uint4 u[2]; v16bf v; };

// ---------------------------------------------------------------------------
// fp32 -> bf16 conversion (weights)
// ---------------------------------------------------------------------------
__global__ void cvt_f32_bf16(const float* __restrict__ src,
                             __bf16* __restrict__ dst, long n) {
    long i = (long)blockIdx.x * 256 + threadIdx.x;
    if (i < n) dst[i] = f2bf(src[i]);
}

// ---------------------------------------------------------------------------
// im2col for 3x3 / pad 1 conv: out[(ci*9 + r*3 + s)*HW + y*W + x]
// Matches W[C,C,3,3] row-major flattened to [C, 2304].
// ---------------------------------------------------------------------------
__global__ void im2col3x3(const float* __restrict__ F, __bf16* __restrict__ col) {
    long idx = (long)blockIdx.x * 256 + threadIdx.x;   // 0 .. 2304*4096
    if (idx >= (long)K9 * HW) return;
    int p  = (int)(idx & (HW - 1));
    int rc = (int)(idx >> 12);
    int s  = rc % 3, r = (rc / 3) % 3, ci = rc / 9;
    int x = p & (WW - 1), y = p >> 6;
    int yy = y + r - 1, xx = x + s - 1;
    float v = 0.f;
    if ((unsigned)yy < (unsigned)HH && (unsigned)xx < (unsigned)WW)
        v = F[(long)ci * HW + yy * WW + xx];
    col[idx] = f2bf(v);
}

// ---------------------------------------------------------------------------
// Staging helpers: global (uint4 = 8 x bf16) -> LDS tile.
//  Mode 1 operand is k-major (one row per k); mode 0 operand is row-major along the
//  tile's own row index (m for A, n for B) -> direct b128 LDS store.
// ---------------------------------------------------------------------------
template<int MODE>
static __device__ __forceinline__ void stage_tile(
    const __bf16* __restrict__ p, long ld, int k0,
    __bf16 (* __restrict__ T)[40], int t_r, int t_c)
{
    if (MODE == 0) {        // tile-row-major source: rows t_r, t_r+64; 8 k's
        #pragma unroll
        for (int h = 0; h < 2; ++h) {
            V8BF t; t.u = *(const uint4*)(p + (long)h * 64 * ld + k0);
            *(uint4*)&T[t_r + h * 64][t_c] = t.u;
        }
    } else {                // k-major source: k rows t_r, t_r+16; 8 tile rows
        #pragma unroll
        for (int h = 0; h < 2; ++h) {
            V8BF t; t.u = *(const uint4*)(p + (long)(k0 + h * 16) * ld);
            #pragma unroll
            for (int i = 0; i < 8; ++i) T[t_c + i][t_r + h * 16] = t.h[i];
        }
    }
}

// ---------------------------------------------------------------------------
// Templated strided bf16 GEMM: C[m,n] = sum_k A(m,k)*B(k,n)
//  AM==0: A(m,k)=A[m*lda+k]   AM==1: A(m,k)=A[k*lda+m]
//  BM==0: B(k,n)=B[k*ldb+n]   BM==1: B(k,n)=B[n*ldb+k]
//  EPI bits: 1=+bias[m]  2=+res[m*ldc+n]  4=store f32  8=store bf16
// LDS rows padded to 40 elems (80 B): rows stay 16B-aligned and 16B accesses
// across 16 lanes (20-bank row stride) hit all 64 banks conflict-free.
// ---------------------------------------------------------------------------
template<int AM, int BM, int EPI>
__global__ __launch_bounds__(256) void gemm_wmma_bf16(
    const __bf16* __restrict__ A, long lda,
    const __bf16* __restrict__ Bm, long ldb,
    int Kdim,
    const float* __restrict__ bias,
    const float* __restrict__ res,
    float*  __restrict__ outF,
    __bf16* __restrict__ outB,
    int ldc)
{
    __shared__ __align__(16) __bf16 As [2][128][40];   // [buf][m][k]
    __shared__ __align__(16) __bf16 Bst[2][128][40];   // [buf][n][k]

    const int tid  = threadIdx.x;
    const int lane = tid & 31;
    const int wave = tid >> 5;
    const int wm   = (wave & 3) * 32;    // wave subtile origin (M)
    const int wn   = (wave >> 2) * 64;   // wave subtile origin (N)
    const long m0  = (long)blockIdx.y * 128;
    const long n0  = (long)blockIdx.x * 128;

    // per-thread staging coordinates (2 chunks of 8 bf16 each)
    const int a_r = (AM == 0) ? (tid >> 2)       : (tid >> 4);
    const int a_c = (AM == 0) ? ((tid & 3) * 8)  : ((tid & 15) * 8);
    const int b_r = (BM == 1) ? (tid >> 2)       : (tid >> 4);
    const int b_c = (BM == 1) ? ((tid & 3) * 8)  : ((tid & 15) * 8);

    const __bf16* pA = (AM == 0) ? (A  + (m0 + a_r) * lda + a_c)
                                 : (A  + (long)a_r * lda + m0 + a_c);
    const __bf16* pB = (BM == 1) ? (Bm + (n0 + b_r) * ldb + b_c)
                                 : (Bm + (long)b_r * ldb + n0 + b_c);

    v8f acc[2][4] = {};

    // prologue: stage first K tile into buffer 0
    stage_tile<AM == 0 ? 0 : 1>(pA, lda, 0, As[0],  a_r, a_c);
    stage_tile<BM == 1 ? 0 : 1>(pB, ldb, 0, Bst[0], b_r, b_c);

    int buf = 0;
    for (int k0 = 0; k0 < Kdim; k0 += 32, buf ^= 1) {
        __syncthreads();     // staged stores for `buf` visible to all waves

        // ---- fragments per ISA 7.12.2 wave32 layouts; 2x b128 LDS each ----
        v16bf af[2], bg[4];
        {
            const int kb = (lane >> 4) * 8;     // A: k in [kb,+8) U [16+kb,+8)
            #pragma unroll
            for (int i = 0; i < 2; ++i) {
                const int mrow = wm + i * 16 + (lane & 15);
                V16BF t;
                t.u[0] = *(const uint4*)&As[buf][mrow][kb];
                t.u[1] = *(const uint4*)&As[buf][mrow][16 + kb];
                af[i] = t.v;
            }
            const int kbb = (lane >> 4) * 8;    // B: k in [kbb,+8) U [16+kbb,+8)
            #pragma unroll
            for (int j = 0; j < 4; ++j) {
                const int ncol = wn + j * 16 + (lane & 15);
                V16BF t;
                t.u[0] = *(const uint4*)&Bst[buf][ncol][kbb];
                t.u[1] = *(const uint4*)&Bst[buf][ncol][16 + kbb];
                bg[j] = t.v;
            }
        }

        #pragma unroll
        for (int i = 0; i < 2; ++i)
            #pragma unroll
            for (int j = 0; j < 4; ++j)
                acc[i][j] = __builtin_amdgcn_wmma_f32_16x16x32_bf16(
                    false, af[i], false, bg[j], (short)0, acc[i][j],
                    false, false);

        // ---- overlap: stage next K tile into other buffer under the WMMAs --
        if (k0 + 32 < Kdim) {
            stage_tile<AM == 0 ? 0 : 1>(pA, lda, k0 + 32, As[buf ^ 1],  a_r, a_c);
            stage_tile<BM == 1 ? 0 : 1>(pB, ldb, k0 + 32, Bst[buf ^ 1], b_r, b_c);
        }
    }

    // ---- epilogue: C/D layout = 8 VGPRs, lanes16-31 carry M+8 ----
    const int nloc  = lane & 15;
    const int mbase = (lane >> 4) * 8;
    #pragma unroll
    for (int i = 0; i < 2; ++i) {
        const long mrow = m0 + wm + i * 16 + mbase;     // 8 consecutive m
        #pragma unroll
        for (int j = 0; j < 4; ++j) {
            const long ng = n0 + wn + j * 16 + nloc;
            #pragma unroll
            for (int r = 0; r < 8; ++r) {
                const long mg = mrow + r;
                float v = acc[i][j][r];
                if (EPI & 1) v += bias[mg];
                if (EPI & 2) v += res[mg * (long)ldc + ng];
                const long o = mg * (long)ldc + ng;
                if (EPI & 4) outF[o] = v;
                if (EPI & 8) outB[o] = f2bf(v);
            }
        }
    }
}

// ---------------------------------------------------------------------------
// Column softmax over S[4096 rows(i) x 4096 cols(k)] (normalize over i, per k)
// ---------------------------------------------------------------------------
__global__ void colsm_stats(const float* __restrict__ S,
                            float* __restrict__ pmax, float* __restrict__ psum) {
    const int k  = blockIdx.x * 256 + threadIdx.x;
    const int ch = blockIdx.y;                       // 16 chunks of 256 rows
    float m = -3.4e38f, sum = 0.f;
    const int i0 = ch * 256;
    for (int i = i0; i < i0 + 256; ++i) {
        float v  = S[(long)i * HW + k];
        float mn = fmaxf(m, v);
        sum = sum * __expf(m - mn) + __expf(v - mn);
        m = mn;
    }
    pmax[ch * HW + k] = m;
    psum[ch * HW + k] = sum;
}

__global__ void colsm_combine(const float* __restrict__ pmax,
                              const float* __restrict__ psum,
                              float* __restrict__ cmax, float* __restrict__ crs) {
    const int k = blockIdx.x * 256 + threadIdx.x;
    float m = -3.4e38f;
    #pragma unroll
    for (int c = 0; c < 16; ++c) m = fmaxf(m, pmax[c * HW + k]);
    float s = 0.f;
    #pragma unroll
    for (int c = 0; c < 16; ++c) s += psum[c * HW + k] * __expf(pmax[c * HW + k] - m);
    cmax[k] = m;
    crs[k]  = 1.f / s;
}

__global__ void colsm_norm(const float* __restrict__ S,
                           const float* __restrict__ cmax,
                           const float* __restrict__ crs,
                           __bf16* __restrict__ P) {
    const long idx = (long)blockIdx.x * 256 + threadIdx.x;
    const int  k   = (int)(idx & (HW - 1));
    P[idx] = f2bf(__expf(S[idx] - cmax[k]) * crs[k]);
}

// ---------------------------------------------------------------------------
// Host orchestration
// ---------------------------------------------------------------------------
template<int AM, int BM, int EPI>
static inline void launch_gemm(hipStream_t s,
                               const __bf16* A, long lda,
                               const __bf16* B, long ldb,
                               int M, int N, int K,
                               const float* bias, const float* res,
                               float* oF, __bf16* oB, int ldc) {
    dim3 grid(N / 128, M / 128);
    gemm_wmma_bf16<AM, BM, EPI><<<grid, 256, 0, s>>>(A, lda, B, ldb, K,
                                                     bias, res, oF, oB, ldc);
}

extern "C" void kernel_launch(void* const* d_in, const int* in_sizes, int n_in,
                              void* d_out, int out_size, void* d_ws, size_t ws_size,
                              hipStream_t stream) {
    (void)in_sizes; (void)n_in; (void)out_size; (void)ws_size;

    // d_in order (setup_inputs dict order):
    // 0:Fl 1:Fv 2:Wqs 3:bqs 4:Wks 5:bks 6:Wvs 7:bvs
    // 8:Wqc 9:bqc 10:Wkc 11:bkc 12:Wvc 13:bvc 14:Wgs 15:bgs 16:Wfuse 17:bfuse
    const float* Fl    = (const float*)d_in[0];
    const float* Fv    = (const float*)d_in[1];
    const float* W3[6] = { (const float*)d_in[2],  (const float*)d_in[4],
                           (const float*)d_in[6],  (const float*)d_in[8],
                           (const float*)d_in[10], (const float*)d_in[12] };
    const float* b3[6] = { (const float*)d_in[3],  (const float*)d_in[5],
                           (const float*)d_in[7],  (const float*)d_in[9],
                           (const float*)d_in[11], (const float*)d_in[13] };
    const float* Wgs   = (const float*)d_in[14];
    const float* bgs   = (const float*)d_in[15];
    const float* Wfuse = (const float*)d_in[16];
    const float* bfuse = (const float*)d_in[17];
    float* out = (float*)d_out;

    // ---- workspace carve-up (~160 MB; all 256B aligned) ----
    char* ws = (char*)d_ws;
    size_t cur = 0;
    auto alloc = [&](size_t bytes) {
        char* p = ws + cur;
        cur += (bytes + 255) & ~(size_t)255;
        return p;
    };
    __bf16* W3b[6];
    for (int i = 0; i < 6; ++i) W3b[i] = (__bf16*)alloc((size_t)CC * K9 * 2);
    __bf16* Wgsb   = (__bf16*)alloc((size_t)CC * CC * 2);
    __bf16* Wfuseb = (__bf16*)alloc((size_t)CC * 2 * CC * 2);
    __bf16* colL   = (__bf16*)alloc((size_t)K9 * HW * 2);
    __bf16* colV   = (__bf16*)alloc((size_t)K9 * HW * 2);
    __bf16* Qb     = (__bf16*)alloc((size_t)CC * HW * 2);
    __bf16* Kb     = (__bf16*)alloc((size_t)CC * HW * 2);
    __bf16* Vb     = (__bf16*)alloc((size_t)CC * HW * 2);
    float*  S      = (float*) alloc((size_t)HW * HW * 4);
    __bf16* P      = (__bf16*)alloc((size_t)HW * HW * 2);
    __bf16* G      = (__bf16*)alloc((size_t)HW * CC * 2);
    __bf16* cat    = (__bf16*)alloc((size_t)2 * CC * HW * 2);
    float*  pmax   = (float*) alloc((size_t)16 * HW * 4);
    float*  psum   = (float*) alloc((size_t)16 * HW * 4);
    float*  cmax   = (float*) alloc((size_t)HW * 4);
    float*  crs    = (float*) alloc((size_t)HW * 4);

    // ---- convert weights to bf16 once ----
    {
        long n = (long)CC * K9;
        for (int i = 0; i < 6; ++i)
            cvt_f32_bf16<<<(n + 255) / 256, 256, 0, stream>>>(W3[i], W3b[i], n);
        long ng = (long)CC * CC;
        cvt_f32_bf16<<<(ng + 255) / 256, 256, 0, stream>>>(Wgs, Wgsb, ng);
        long nf = (long)CC * 2 * CC;
        cvt_f32_bf16<<<(nf + 255) / 256, 256, 0, stream>>>(Wfuse, Wfuseb, nf);
    }

    const long imgN = (long)K9 * HW;

    for (int b = 0; b < 2; ++b) {
        const float* Fl_b = Fl + (size_t)b * CC * HW;
        const float* Fv_b = Fv + (size_t)b * CC * HW;

        im2col3x3<<<(imgN + 255) / 256, 256, 0, stream>>>(Fl_b, colL);
        im2col3x3<<<(imgN + 255) / 256, 256, 0, stream>>>(Fv_b, colV);

        // branch 0 = self (Q,K,V from Fl, residual Fl)
        // branch 1 = cross (Q from Fl; K,V from Fv, residual Fv)
        for (int br = 0; br < 2; ++br) {
            const __bf16* wq = W3b[br ? 3 : 0];
            const __bf16* wk = W3b[br ? 4 : 1];
            const __bf16* wv = W3b[br ? 5 : 2];
            const float*  bq = b3[br ? 3 : 0];
            const float*  bk = b3[br ? 4 : 1];
            const float*  bv = b3[br ? 5 : 2];
            const __bf16* colQ  = colL;              // Q always from Fl
            const __bf16* colKV = br ? colV : colL;
            const float*  resid = br ? Fv_b : Fl_b;

            // 3x3 convs as implicit GEMMs: [256 x 2304] x [2304 x 4096]
            launch_gemm<0,0,9>(stream, wq, K9, colQ,  HW, CC, HW, K9,
                               bq, nullptr, nullptr, Qb, HW);
            launch_gemm<0,0,9>(stream, wk, K9, colKV, HW, CC, HW, K9,
                               bk, nullptr, nullptr, Kb, HW);
            launch_gemm<0,0,9>(stream, wv, K9, colKV, HW, CC, HW, K9,
                               bv, nullptr, nullptr, Vb, HW);

            // S[i,k] = sum_c Q[c,i] K[c,k]   (A col-major, B row-major)
            launch_gemm<1,0,4>(stream, Qb, HW, Kb, HW, HW, HW, CC,
                               nullptr, nullptr, S, nullptr, HW);

            // softmax over i (columns of S), write P bf16 row-major [i,k]
            colsm_stats  <<<dim3(HW / 256, 16), 256, 0, stream>>>(S, pmax, psum);
            colsm_combine<<<HW / 256, 256, 0, stream>>>(pmax, psum, cmax, crs);
            colsm_norm   <<<(long)HW * HW / 256, 256, 0, stream>>>(S, cmax, crs, P);

            // G[i,j] = sum_k P[i,k] V[j,k]   (A row-major, B col-major)
            // flat [i,j] row-major (ldc=256) IS the torch "raw reshape" [C,H,W]
            launch_gemm<0,1,8>(stream, P, HW, Vb, HW, HW, CC, HW,
                               nullptr, nullptr, nullptr, G, CC);

            // 1x1 conv (shared Wgs/bgs) + residual -> concat row block
            launch_gemm<0,0,11>(stream, Wgsb, CC,
                                G /* reinterpreted [256,4096] */, HW,
                                CC, HW, CC, bgs, resid,
                                nullptr, cat + (size_t)br * CC * HW, HW);
        }

        // fuse: [256 x 512] x [512 x 4096] + bias -> fp32 output
        launch_gemm<0,0,5>(stream, Wfuseb, 2 * CC, cat, HW,
                           CC, HW, 2 * CC, bfuse, nullptr,
                           out + (size_t)b * CC * HW, nullptr, HW);
    }
}
